// Decoder_49950469652964
// MI455X (gfx1250) — compile-verified, matrix operand order checked by reading, not run on hardware
//
#include <hip/hip_runtime.h>
#include <hip/hip_bf16.h>
#include <math.h>

typedef __bf16 bf16;
typedef __attribute__((ext_vector_type(16))) __bf16        v16bf;
typedef __attribute__((ext_vector_type(8)))  float         v8f;
typedef __attribute__((ext_vector_type(4)))  unsigned int  u32x4;

#define VOCAB  32000
#define EMB    512
#define HID    1024
#define BATCH  32
#define TSTEPS 32
#define ENC_S  64
#define SOS    1

// ---------------------------------------------------------------------------
// WMMA fragment load (bf16, 16x32 A-tile / 32x16 B-tile from [rows,K] row-major)
// lane 0-15: row = lane,    K chunk {kb..kb+7, kb+16..kb+23} with kb = 0
// lane16-31: row = lane-16, same pattern with kb = 8
// caller passes pointer already offset to (row, k0 + kb); we grab 2x16 bytes.
// ---------------------------------------------------------------------------
__device__ __forceinline__ v16bf load_frag_bf16(const bf16* p) {
    union { v16bf v; u32x4 u[2]; } r;
    r.u[0] = *(const u32x4*)(p);
    r.u[1] = *(const u32x4*)(p + 16);
    return r.v;
}

__device__ __forceinline__ v8f wmma_bf16(v16bf a, v16bf b, v8f c) {
    return __builtin_amdgcn_wmma_f32_16x16x32_bf16(false, a, false, b,
                                                   (short)0, c, false, false);
}

// ---------------------------------------------------------------------------
// Small GEMM: C[32,N] = A[32,K](bf16) @ W[N,K]^T(bf16) + bias, f32 out.
// Block = 256 threads = 8 waves; wave w -> columns [blk*128 + w*16, +16).
// ---------------------------------------------------------------------------
__global__ void gemm32_bias_kernel(const bf16* __restrict__ A,
                                   const bf16* __restrict__ W,
                                   const float* __restrict__ bias,
                                   float* __restrict__ C, int N, int K) {
    int lane = threadIdx.x & 31;
    int wave = threadIdx.x >> 5;
    int nt   = blockIdx.x * 128 + wave * 16;
    int m    = lane & 15;
    int kb   = (lane >> 4) << 3;
    v8f acc0 = {0.f,0.f,0.f,0.f,0.f,0.f,0.f,0.f};
    v8f acc1 = acc0;
    const bf16* a0p = A + (size_t)m * K + kb;
    const bf16* a1p = A + (size_t)(m + 16) * K + kb;
    const bf16* wp  = W + (size_t)(nt + m) * K + kb;
    for (int k = 0; k < K; k += 32) {
        v16bf a0 = load_frag_bf16(a0p + k);
        v16bf a1 = load_frag_bf16(a1p + k);
        v16bf b  = load_frag_bf16(wp + k);
        acc0 = wmma_bf16(a0, b, acc0);
        acc1 = wmma_bf16(a1, b, acc1);
    }
    int col   = nt + (lane & 15);
    int rbase = (lane >> 4) << 3;
    float bv  = bias[col];
#pragma unroll
    for (int r = 0; r < 8; ++r) {
        int row = rbase + r;
        C[(size_t)row * N + col]        = acc0[r] + bv;
        C[(size_t)(row + 16) * N + col] = acc1[r] + bv;
    }
}

// ---------------------------------------------------------------------------
// Small GEMM + tanh, bf16 out: O[row_off+m, n] = tanh(A[32,K] @ W[N,K]^T)
// ---------------------------------------------------------------------------
__global__ void gemm32_tanh_kernel(const bf16* __restrict__ A,
                                   const bf16* __restrict__ W,
                                   bf16* __restrict__ O, int N, int K,
                                   int row_off) {
    int lane = threadIdx.x & 31;
    int wave = threadIdx.x >> 5;
    int nt   = blockIdx.x * 128 + wave * 16;
    int m    = lane & 15;
    int kb   = (lane >> 4) << 3;
    v8f acc0 = {0.f,0.f,0.f,0.f,0.f,0.f,0.f,0.f};
    v8f acc1 = acc0;
    const bf16* a0p = A + (size_t)m * K + kb;
    const bf16* a1p = A + (size_t)(m + 16) * K + kb;
    const bf16* wp  = W + (size_t)(nt + m) * K + kb;
    for (int k = 0; k < K; k += 32) {
        v16bf a0 = load_frag_bf16(a0p + k);
        v16bf a1 = load_frag_bf16(a1p + k);
        v16bf b  = load_frag_bf16(wp + k);
        acc0 = wmma_bf16(a0, b, acc0);
        acc1 = wmma_bf16(a1, b, acc1);
    }
    int col   = nt + (lane & 15);
    int rbase = (lane >> 4) << 3;
#pragma unroll
    for (int r = 0; r < 8; ++r) {
        int row = rbase + r;
        O[(size_t)(row_off + row) * N + col]      = (bf16)tanhf(acc0[r]);
        O[(size_t)(row_off + row + 16) * N + col] = (bf16)tanhf(acc1[r]);
    }
}

// ---------------------------------------------------------------------------
// Big fc GEMM: logits = o_all[1024,1024] @ fc_w[V,1024]^T + fc_b,
// written straight into d_out at [b][t][v] (row = t*32+b).
// Block = 8 waves as 2(M) x 4(N); wave tile 32x64:
//   per K-step: 2 A-frags + 4 B-frags (12 x b128 loads) feed 8 WMMAs.
// Grid = (V/256, 1024/64) = (125, 16).
// ---------------------------------------------------------------------------
__global__ void fc_gemm_kernel(const bf16* __restrict__ A,
                               const bf16* __restrict__ W,
                               const float* __restrict__ bias,
                               float* __restrict__ Out) {
    int lane  = threadIdx.x & 31;
    int wave  = threadIdx.x >> 5;
    int mtile = blockIdx.y * 64 + (wave >> 2) * 32;
    int ntile = blockIdx.x * 256 + (wave & 3) * 64;
    int m     = lane & 15;
    int kb    = (lane >> 4) << 3;
    const int K = HID;
    const bf16* a0p = A + (size_t)(mtile + m) * K + kb;
    const bf16* a1p = a0p + (size_t)16 * K;
    const bf16* b0p = W + (size_t)(ntile + m) * K + kb;
    const bf16* b1p = b0p + (size_t)16 * K;
    const bf16* b2p = b0p + (size_t)32 * K;
    const bf16* b3p = b0p + (size_t)48 * K;
    v8f z = {0.f,0.f,0.f,0.f,0.f,0.f,0.f,0.f};
    v8f acc00 = z, acc01 = z, acc02 = z, acc03 = z;
    v8f acc10 = z, acc11 = z, acc12 = z, acc13 = z;
    for (int k = 0; k < K; k += 32) {
        v16bf a0 = load_frag_bf16(a0p + k);
        v16bf a1 = load_frag_bf16(a1p + k);
        v16bf b0 = load_frag_bf16(b0p + k);
        v16bf b1 = load_frag_bf16(b1p + k);
        v16bf b2 = load_frag_bf16(b2p + k);
        v16bf b3 = load_frag_bf16(b3p + k);
        acc00 = wmma_bf16(a0, b0, acc00);
        acc01 = wmma_bf16(a0, b1, acc01);
        acc02 = wmma_bf16(a0, b2, acc02);
        acc03 = wmma_bf16(a0, b3, acc03);
        acc10 = wmma_bf16(a1, b0, acc10);
        acc11 = wmma_bf16(a1, b1, acc11);
        acc12 = wmma_bf16(a1, b2, acc12);
        acc13 = wmma_bf16(a1, b3, acc13);
    }
    int rbase = (lane >> 4) << 3;
    int cloc  = lane & 15;
    int c0 = ntile + cloc;
    int c1 = c0 + 16;
    int c2 = c0 + 32;
    int c3 = c0 + 48;
    float bv0 = bias[c0], bv1 = bias[c1], bv2 = bias[c2], bv3 = bias[c3];
#pragma unroll
    for (int r = 0; r < 8; ++r) {
        int r0 = mtile + rbase + r;       // global GEMM row (= t*32 + b)
        int r1 = r0 + 16;
        size_t o0 = (size_t)(r0 & 31) * (TSTEPS * VOCAB) + (size_t)(r0 >> 5) * VOCAB;
        size_t o1 = (size_t)(r1 & 31) * (TSTEPS * VOCAB) + (size_t)(r1 >> 5) * VOCAB;
        Out[o0 + c0] = acc00[r] + bv0;
        Out[o0 + c1] = acc01[r] + bv1;
        Out[o0 + c2] = acc02[r] + bv2;
        Out[o0 + c3] = acc03[r] + bv3;
        Out[o1 + c0] = acc10[r] + bv0;
        Out[o1 + c1] = acc11[r] + bv1;
        Out[o1 + c2] = acc12[r] + bv2;
        Out[o1 + c3] = acc13[r] + bv3;
    }
}

// ---------------------------------------------------------------------------
// GRU gate nonlinearity + Luong attention, one block per batch element.
// ---------------------------------------------------------------------------
__global__ void gru_attn_kernel(const float* __restrict__ gi,
                                const float* __restrict__ gh,
                                float* __restrict__ h, bf16* __restrict__ hbf,
                                const float* __restrict__ enc,
                                bf16* __restrict__ catbf) {
    __shared__ float hs[HID];
    __shared__ float sc[ENC_S];
    __shared__ float wsm[ENC_S];
    int b   = blockIdx.x;
    int tid = threadIdx.x;
    const float* gib = gi + (size_t)b * 3 * HID;
    const float* ghb = gh + (size_t)b * 3 * HID;
    for (int i = tid; i < HID; i += blockDim.x) {
        float r  = 1.f / (1.f + __expf(-(gib[i] + ghb[i])));
        float z  = 1.f / (1.f + __expf(-(gib[HID + i] + ghb[HID + i])));
        float n  = tanhf(gib[2 * HID + i] + r * ghb[2 * HID + i]);
        float hp = h[(size_t)b * HID + i];
        float hn = (1.f - z) * n + z * hp;
        hs[i] = hn;
        h[(size_t)b * HID + i]          = hn;
        hbf[(size_t)b * HID + i]        = (bf16)hn;
        catbf[(size_t)b * 2 * HID + i]  = (bf16)hn;
    }
    __syncthreads();
    int wave = tid >> 5, lane = tid & 31;
    const float* encb = enc + (size_t)b * ENC_S * HID;
    for (int s = wave * 8; s < wave * 8 + 8; ++s) {
        float p = 0.f;
        for (int j = lane; j < HID; j += 32) p += encb[(size_t)s * HID + j] * hs[j];
        for (int off = 16; off > 0; off >>= 1) p += __shfl_xor(p, off, 32);
        if (lane == 0) sc[s] = p;
    }
    __syncthreads();
    if (tid == 0) {
        float mx = sc[0];
        for (int s = 1; s < ENC_S; ++s) mx = fmaxf(mx, sc[s]);
        float sum = 0.f;
        for (int s = 0; s < ENC_S; ++s) { float e = __expf(sc[s] - mx); wsm[s] = e; sum += e; }
        float inv = 1.f / sum;
        for (int s = 0; s < ENC_S; ++s) wsm[s] *= inv;
    }
    __syncthreads();
    for (int i = tid; i < HID; i += blockDim.x) {
        float acc = 0.f;
        for (int s = 0; s < ENC_S; ++s) acc += wsm[s] * encb[(size_t)s * HID + i];
        catbf[(size_t)b * 2 * HID + HID + i] = (bf16)acc;
    }
}

// ---------------------------------------------------------------------------
// Helpers: fp32 -> bf16 convert, embedding gather, h init, log-softmax, copy.
// ---------------------------------------------------------------------------
__global__ void f2bf_kernel(const float* __restrict__ src, bf16* __restrict__ dst, int n) {
    int i = blockIdx.x * blockDim.x + threadIdx.x;
    if (i < n) dst[i] = (bf16)src[i];
}

__global__ void embed_kernel(const int* __restrict__ target,
                             const float* __restrict__ emb,
                             bf16* __restrict__ xbf) {
    int blk = blockIdx.x;           // = t*BATCH + b
    int t = blk >> 5, b = blk & 31; // BATCH == 32
    int id = (t == 0) ? SOS : target[b * TSTEPS + (t - 1)];
    const float* src = emb + (size_t)id * EMB;
    bf16* dst = xbf + (size_t)blk * EMB;
    for (int e = threadIdx.x; e < EMB; e += blockDim.x) dst[e] = (bf16)src[e];
}

__global__ void hinit_kernel(const float* __restrict__ eh,
                             float* __restrict__ h, bf16* __restrict__ hbf) {
    int i = blockIdx.x * blockDim.x + threadIdx.x;
    float v = eh[i];
    h[i] = v;
    hbf[i] = (bf16)v;
}

__global__ void logsoftmax_kernel(float* __restrict__ out) {
    __shared__ float red[256];
    float* row = out + (size_t)blockIdx.x * VOCAB;
    int tid = threadIdx.x;
    float mx = -3.402823466e+38f;
    for (int i = tid; i < VOCAB; i += 256) mx = fmaxf(mx, row[i]);
    red[tid] = mx; __syncthreads();
    for (int s = 128; s > 0; s >>= 1) {
        if (tid < s) red[tid] = fmaxf(red[tid], red[tid + s]);
        __syncthreads();
    }
    mx = red[0]; __syncthreads();
    float sum = 0.f;
    for (int i = tid; i < VOCAB; i += 256) sum += __expf(row[i] - mx);
    red[tid] = sum; __syncthreads();
    for (int s = 128; s > 0; s >>= 1) {
        if (tid < s) red[tid] += red[tid + s];
        __syncthreads();
    }
    float lse = mx + __logf(red[0]);
    for (int i = tid; i < VOCAB; i += 256) row[i] = row[i] - lse;
}

__global__ void copy_hidden_kernel(const float* __restrict__ h, float* __restrict__ dst) {
    int i = blockIdx.x * blockDim.x + threadIdx.x;
    dst[i] = h[i];
}

// ---------------------------------------------------------------------------
extern "C" void kernel_launch(void* const* d_in, const int* in_sizes, int n_in,
                              void* d_out, int out_size, void* d_ws, size_t ws_size,
                              hipStream_t stream) {
    (void)in_sizes; (void)n_in; (void)out_size; (void)ws_size;
    const int*   target = (const int*)  d_in[0];
    const float* enc_h  = (const float*)d_in[1];
    const float* enc_o  = (const float*)d_in[2];
    const float* emb    = (const float*)d_in[3];
    const float* w_ih   = (const float*)d_in[4];
    const float* w_hh   = (const float*)d_in[5];
    const float* b_ih   = (const float*)d_in[6];
    const float* b_hh   = (const float*)d_in[7];
    const float* wa     = (const float*)d_in[8];
    const float* fc_w   = (const float*)d_in[9];
    const float* fc_b   = (const float*)d_in[10];

    char* ws = (char*)d_ws;
    auto carve = [&](size_t bytes) {
        char* p = ws;
        ws += (bytes + 255) & ~(size_t)255;
        return p;
    };
    bf16*  fcw_bf = (bf16*) carve((size_t)VOCAB * HID * 2);        // 65.5 MB
    bf16*  wih_bf = (bf16*) carve((size_t)3 * HID * EMB * 2);
    bf16*  whh_bf = (bf16*) carve((size_t)3 * HID * HID * 2);
    bf16*  wa_bf  = (bf16*) carve((size_t)HID * 2 * HID * 2);
    bf16*  x_bf   = (bf16*) carve((size_t)TSTEPS * BATCH * EMB * 2);
    bf16*  o_bf   = (bf16*) carve((size_t)TSTEPS * BATCH * HID * 2);
    bf16*  h_bf   = (bf16*) carve((size_t)BATCH * HID * 2);
    bf16*  cat_bf = (bf16*) carve((size_t)BATCH * 2 * HID * 2);
    float* gi     = (float*)carve((size_t)BATCH * 3 * HID * 4);
    float* gh     = (float*)carve((size_t)BATCH * 3 * HID * 4);
    float* h      = (float*)carve((size_t)BATCH * HID * 4);

    // -------- one-time precompute (per launch) --------
    f2bf_kernel<<<(VOCAB * HID + 255) / 256, 256, 0, stream>>>(fc_w, fcw_bf, VOCAB * HID);
    f2bf_kernel<<<(3 * HID * EMB + 255) / 256, 256, 0, stream>>>(w_ih, wih_bf, 3 * HID * EMB);
    f2bf_kernel<<<(3 * HID * HID + 255) / 256, 256, 0, stream>>>(w_hh, whh_bf, 3 * HID * HID);
    f2bf_kernel<<<(HID * 2 * HID + 255) / 256, 256, 0, stream>>>(wa, wa_bf, HID * 2 * HID);
    embed_kernel<<<TSTEPS * BATCH, 256, 0, stream>>>(target, emb, x_bf);
    hinit_kernel<<<(BATCH * HID) / 256, 256, 0, stream>>>(enc_h, h, h_bf);

    // -------- serial GRU + attention (T steps) --------
    for (int t = 0; t < TSTEPS; ++t) {
        const bf16* xt = x_bf + (size_t)t * BATCH * EMB;
        gemm32_bias_kernel<<<3 * HID / 128, 256, 0, stream>>>(xt, wih_bf, b_ih, gi, 3 * HID, EMB);
        gemm32_bias_kernel<<<3 * HID / 128, 256, 0, stream>>>(h_bf, whh_bf, b_hh, gh, 3 * HID, HID);
        gru_attn_kernel<<<BATCH, 256, 0, stream>>>(gi, gh, h, h_bf, enc_o, cat_bf);
        gemm32_tanh_kernel<<<HID / 128, 256, 0, stream>>>(cat_bf, wa_bf, o_bf, HID, 2 * HID, t * BATCH);
    }

    // -------- big fused fc projection over all T at once --------
    dim3 fcgrid(VOCAB / 256, (TSTEPS * BATCH) / 64);
    fc_gemm_kernel<<<fcgrid, 256, 0, stream>>>(o_bf, fcw_bf, fc_b, (float*)d_out);

    // -------- in-place log-softmax on d_out rows --------
    logsoftmax_kernel<<<TSTEPS * BATCH, 256, 0, stream>>>((float*)d_out);

    // -------- final hidden state --------
    copy_hidden_kernel<<<(BATCH * HID) / 256, 256, 0, stream>>>(
        h, (float*)d_out + (size_t)BATCH * TSTEPS * VOCAB);
}
